// ImpactModel_87737591922736
// MI455X (gfx1250) — matrix operand order repaired and checked by reading, not compiled, not run on hardware
//
#include <hip/hip_runtime.h>
#include <math.h>

typedef __attribute__((ext_vector_type(16))) _Float16 v16h;
typedef __attribute__((ext_vector_type(8)))  _Float16 v8h;
typedef __attribute__((ext_vector_type(8)))  float    v8f;
typedef __attribute__((ext_vector_type(4)))  float    v4f;

#define T_SNAP 10
#define NPAP   30000
#define FDIM   256
#define HDIM   128
#define LH     5
#define NSTEPS 5
#define NYR    5

#define ROWS_PER_WAVE 16
#define WAVES_PER_WG  8
#define ROWS_PER_WG   (ROWS_PER_WAVE * WAVES_PER_WG)           // 128
#define NUM_WG        ((NPAP + ROWS_PER_WG - 1) / ROWS_PER_WG) // 235

#define FRAG_HALFS 512   // 32 lanes * 16 halfs per WMMA B fragment

// ---- workspace byte offsets (prepped f16 fragment blobs) ----
#define WS_PARTIALS 0
#define WS_WENC     4096                         // 8*8   frags = 65536 B
#define WS_WIH      (WS_WENC + 65536)            // 4*24  frags = 98304 B
#define WS_WHH      (WS_WIH + 98304)             // 4*24  frags = 98304 B
#define WS_WHEADS   (WS_WHH + 98304)             // 4*1   frags =  4096 B

__device__ __forceinline__ v8f wmma_f16(v16h a, v16h b, v8f c) {
  return __builtin_amdgcn_wmma_f32_16x16x32_f16(false, a, false, b, (short)0, c,
                                                false, false);
}

__device__ __forceinline__ v8f zero8() {
  v8f z;
  #pragma unroll
  for (int r = 0; r < 8; ++r) z[r] = 0.f;
  return z;
}

__device__ __forceinline__ float softplusf(float x) {
  return (x > 20.f) ? x : log1pf(expf(x));
}
__device__ __forceinline__ float sigmoidf(float x) {
  return 1.f / (1.f + expf(-x));
}
__device__ __forceinline__ float normcdff_(float z) {
  return 0.5f * (1.0f + erff(z * 0.70710678118654752f));
}

// Load an A fragment (16x32 f16) from a row-major [16 x 128] f16 tile.
__device__ __forceinline__ v16h load_a_frag(const _Float16* tile, int kk,
                                            int nlo, int half) {
  const _Float16* p = tile + nlo * HDIM + kk * 32 + half * 8;
  v8h a0 = *(const v8h*)(p);
  v8h a1 = *(const v8h*)(p + 16);
  return __builtin_shufflevector(a0, a1, 0, 1, 2, 3, 4, 5, 6, 7, 8, 9, 10, 11,
                                 12, 13, 14, 15);
}

// Load a pre-swizzled B fragment (32 contiguous bytes per lane).
__device__ __forceinline__ v16h load_b_frag(const _Float16* base, int frag,
                                            int lane) {
  return *(const v16h*)(base + (size_t)frag * FRAG_HALFS + lane * 16);
}

// Pack a f32 C-fragment to 8 halfs (one ds_store_b128 per lane).
__device__ __forceinline__ v8h pack_c(v8f c) {
  v8h o;
  #pragma unroll
  for (int r = 0; r < 8; ++r) o[r] = (_Float16)c[r];
  return o;
}

// -------- prep: convert f32 weight [K x ncols] into WMMA-B fragment order ----
__global__ void prep_w(const float* __restrict__ W, _Float16* __restrict__ dst,
                       int ntiles, int ncols) {
  int f = blockIdx.x;
  int kk = f / ntiles, nn = f % ntiles;
  int lane = threadIdx.x;
  int hh = lane >> 4, n = lane & 15;
  int col = nn * 16 + n;
  _Float16* o = dst + (size_t)f * FRAG_HALFS + lane * 16;
  #pragma unroll
  for (int i = 0; i < 16; ++i) {
    int K = kk * 32 + (i < 8 ? hh * 8 + i : 16 + hh * 8 + (i - 8));
    float v = (col < ncols) ? W[(size_t)K * ncols + col] : 0.f;
    o[i] = (_Float16)v;
  }
}

// ---------------- fused encoder + GRU + heads + loss kernel ------------------
__launch_bounds__(256)
__global__ void impact_fused(const float* __restrict__ X,
                             const float* __restrict__ b_ih,
                             const float* __restrict__ b_hh,
                             const float* __restrict__ b_heads,
                             const float* __restrict__ Y,
                             const int*   __restrict__ years,
                             const _Float16* __restrict__ wencF,
                             const _Float16* __restrict__ wihF,
                             const _Float16* __restrict__ whhF,
                             const _Float16* __restrict__ wheadsF,
                             float* __restrict__ partials) {
  __shared__ __align__(32) _Float16 sWenc[8 * 8 * FRAG_HALFS];     // 64 KB
  __shared__ __align__(32) _Float16 sWhh[4 * 24 * FRAG_HALFS];     // 96 KB
  __shared__ __align__(32) _Float16 sWhd[4 * FRAG_HALFS];          //  4 KB
  // Per-wave row-major [16 x 128] f16 tile: first the emb A-tile, then h.
  __shared__ __align__(32) _Float16 sH[WAVES_PER_WG][ROWS_PER_WAVE * HDIM]; // 32 KB
  // gx in C-fragment order: [frag 0..23][lane][8 halfs]  -> 12 KB/wave
  __shared__ __align__(32) _Float16 sGx[WAVES_PER_WG][24 * 32 * 8]; // 96 KB
  __shared__ float sHeads[WAVES_PER_WG][ROWS_PER_WAVE * 4];        //  2 KB
  __shared__ float sRed[WAVES_PER_WG][2];

  const int tid = threadIdx.x;

  // Cooperative copy of W_enc / W_hh / W_heads fragments into LDS.
  {
    const uint4* s1 = (const uint4*)wencF;
    uint4* d1 = (uint4*)sWenc;
    const int n1 = 8 * 8 * FRAG_HALFS * 2 / 16;
    for (int i = tid; i < n1; i += 256) d1[i] = s1[i];
    const uint4* s2 = (const uint4*)whhF;
    uint4* d2 = (uint4*)sWhh;
    const int n2 = 4 * 24 * FRAG_HALFS * 2 / 16;
    for (int i = tid; i < n2; i += 256) d2[i] = s2[i];
    const uint4* s3 = (const uint4*)wheadsF;
    uint4* d3 = (uint4*)sWhd;
    const int n3 = 4 * FRAG_HALFS * 2 / 16;
    for (int i = tid; i < n3; i += 256) d3[i] = s3[i];
  }
  __syncthreads();

  const int wave = tid >> 5, lane = tid & 31;
  const int half = lane >> 4, nlo = lane & 15;
  const int p0 = (blockIdx.x * WAVES_PER_WG + wave) * ROWS_PER_WAVE;

  float accTime = 0.f, accPred = 0.f;
  v8f embPrev[8];
  #pragma unroll
  for (int nn = 0; nn < 8; ++nn) embPrev[nn] = zero8();

  for (int t = 0; t < T_SNAP; ++t) {
    // ---------------- encoder: emb = relu(X_t @ W_enc) ----------------------
    v8f emb[8];
    #pragma unroll
    for (int nn = 0; nn < 8; ++nn) emb[nn] = zero8();

    #pragma unroll
    for (int kk = 0; kk < 8; ++kk) {
      // A fragment straight from global X (f32 -> f16), rows OOB -> 0.
      v16h a;
      int p = p0 + nlo;
      if (p < NPAP) {
        const float* xp = X + ((size_t)t * NPAP + p) * FDIM + kk * 32 + half * 8;
        v4f x0 = *(const v4f*)(xp);
        v4f x1 = *(const v4f*)(xp + 4);
        v4f x2 = *(const v4f*)(xp + 16);
        v4f x3 = *(const v4f*)(xp + 20);
        #pragma unroll
        for (int i = 0; i < 4; ++i) {
          a[i]      = (_Float16)x0[i];
          a[4 + i]  = (_Float16)x1[i];
          a[8 + i]  = (_Float16)x2[i];
          a[12 + i] = (_Float16)x3[i];
        }
      } else {
        #pragma unroll
        for (int i = 0; i < 16; ++i) a[i] = (_Float16)0.f;
      }
      #pragma unroll
      for (int nn = 0; nn < 8; ++nn) {
        v16h b = load_b_frag(sWenc, kk * 8 + nn, lane);
        emb[nn] = wmma_f16(a, b, emb[nn]);
      }
    }
    #pragma unroll
    for (int nn = 0; nn < 8; ++nn)
      #pragma unroll
      for (int r = 0; r < 8; ++r) emb[nn][r] = fmaxf(emb[nn][r], 0.f);

    // ---------------- temporal smoothness ----------------------------------
    if (t > 0) {
      #pragma unroll
      for (int nn = 0; nn < 8; ++nn)
        #pragma unroll
        for (int r = 0; r < 8; ++r) {
          float d = embPrev[nn][r] - emb[nn][r];
          accTime += d * d;
        }
    }
    #pragma unroll
    for (int nn = 0; nn < 8; ++nn) embPrev[nn] = emb[nn];

    // ---------------- GRU + heads + loss for training years -----------------
    bool inY = false;
    #pragma unroll
    for (int q = 0; q < NYR; ++q) inY = inY || (years[q] == t);
    if (inY) {
      // Stage emb as f16 A-tile in the (currently free) h buffer.
      _Float16* ht = sH[wave];
      #pragma unroll
      for (int nn = 0; nn < 8; ++nn)
        #pragma unroll
        for (int r = 0; r < 8; ++r)
          ht[(r + 8 * half) * HDIM + nn * 16 + nlo] = (_Float16)emb[nn][r];

      // gx = emb @ W_ih + b_ih; B streamed from L2, result -> f16 frags in LDS.
      {
        v16h eA[4];
        #pragma unroll
        for (int kk = 0; kk < 4; ++kk) eA[kk] = load_a_frag(ht, kk, nlo, half);
        #pragma unroll
        for (int nn = 0; nn < 24; ++nn) {
          v8f c;
          float bv = b_ih[nn * 16 + nlo];
          #pragma unroll
          for (int r = 0; r < 8; ++r) c[r] = bv;
          #pragma unroll
          for (int kk = 0; kk < 4; ++kk)
            c = wmma_f16(eA[kk], load_b_frag(wihF, kk * 24 + nn, lane), c);
          *(v8h*)(sGx[wave] + (nn * 32 + lane) * 8) = pack_c(c);
        }
      }

      // h0 = 0 (emb tile no longer needed; l_time kept in registers).
      v8f hC[8];
      #pragma unroll
      for (int j = 0; j < 8; ++j) hC[j] = zero8();
      {
        unsigned int* hz = (unsigned int*)sH[wave];
        for (int i = lane; i < ROWS_PER_WAVE * HDIM / 2; i += 32) hz[i] = 0u;
      }

      for (int s = 0; s < NSTEPS; ++s) {
        // A-fragments of h for this step (loaded before any h writes).
        v16h hA[4];
        #pragma unroll
        for (int kk = 0; kk < 4; ++kk) hA[kk] = load_a_frag(ht, kk, nlo, half);

        #pragma unroll
        for (int j = 0; j < 8; ++j) {
          v8f gr, gz, gn;
          float bvr = b_hh[j * 16 + nlo];
          float bvz = b_hh[128 + j * 16 + nlo];
          float bvn = b_hh[256 + j * 16 + nlo];
          #pragma unroll
          for (int r = 0; r < 8; ++r) { gr[r] = bvr; gz[r] = bvz; gn[r] = bvn; }
          #pragma unroll
          for (int kk = 0; kk < 4; ++kk) {
            gr = wmma_f16(hA[kk], load_b_frag(sWhh, kk * 24 + j, lane), gr);
            gz = wmma_f16(hA[kk], load_b_frag(sWhh, kk * 24 + j + 8, lane), gz);
            gn = wmma_f16(hA[kk], load_b_frag(sWhh, kk * 24 + j + 16, lane), gn);
          }
          v8h xr = *(const v8h*)(sGx[wave] + ((j)      * 32 + lane) * 8);
          v8h xz = *(const v8h*)(sGx[wave] + ((j + 8)  * 32 + lane) * 8);
          v8h xn = *(const v8h*)(sGx[wave] + ((j + 16) * 32 + lane) * 8);
          #pragma unroll
          for (int r = 0; r < 8; ++r) {
            float rr = sigmoidf((float)xr[r] + gr[r]);
            float zz = sigmoidf((float)xz[r] + gz[r]);
            float nv = tanhf((float)xn[r] + rr * gn[r]);
            float hv = (1.f - zz) * nv + zz * hC[j][r];
            hC[j][r] = hv;
            // Safe to overwrite: this step's A-frags are already in hA[].
            ht[(r + 8 * half) * HDIM + j * 16 + nlo] = (_Float16)hv;
          }
        }
      }

      // heads = h_last @ W_heads(+pad) + b_heads
      {
        v8f hd;
        float bh = (nlo < 3) ? b_heads[nlo] : 0.f;
        #pragma unroll
        for (int r = 0; r < 8; ++r) hd[r] = bh;
        #pragma unroll
        for (int kk = 0; kk < 4; ++kk) {
          v16h a = load_a_frag(ht, kk, nlo, half);
          hd = wmma_f16(a, load_b_frag(sWhd, kk, lane), hd);
        }
        if (nlo < 3) {
          #pragma unroll
          for (int r = 0; r < 8; ++r)
            sHeads[wave][(r + 8 * half) * 4 + nlo] = hd[r];
        }
      }
      // Same-wave LDS ops are in order: reads below observe the stores above.

      const float* hbuf = sHeads[wave];
      for (int it = lane; it < ROWS_PER_WAVE * LH; it += 32) {
        int row = it & 15;
        int l = it >> 4;  // horizon index 0..4 (horizon value l+1)
        int p = p0 + row;
        if (p < NPAP) {
          float eta = softplusf(hbuf[row * 4 + 0]);
          float mu = hbuf[row * 4 + 1];
          float sg = softplusf(hbuf[row * 4 + 2]) + 1e-3f;
          float c1 = eta * normcdff_((logf((float)(l + 1)) - mu) / sg);
          float yhat;
          if (l == 0) {
            yhat = c1;
          } else {
            float c0 = eta * normcdff_((logf((float)l) - mu) / sg);
            yhat = c1 - c0;
          }
          float yt = Y[((size_t)t * NPAP + p) * LH + l];
          float d = log1pf(yt + 1.0f) - log1pf(yhat);
          accPred += d * d;
        }
      }
    }
  }

  // ---------------- reduction: wave -> WG -> partials -----------------------
  #pragma unroll
  for (int off = 16; off > 0; off >>= 1) {
    accPred += __shfl_xor(accPred, off, 32);
    accTime += __shfl_xor(accTime, off, 32);
  }
  if (lane == 0) { sRed[wave][0] = accPred; sRed[wave][1] = accTime; }
  __syncthreads();
  if (tid == 0) {
    float sp = 0.f, st = 0.f;
    #pragma unroll
    for (int w = 0; w < WAVES_PER_WG; ++w) { sp += sRed[w][0]; st += sRed[w][1]; }
    partials[blockIdx.x * 2 + 0] = sp;
    partials[blockIdx.x * 2 + 1] = st;
  }
}

__global__ void finalize_loss(const float* __restrict__ partials,
                              float* __restrict__ out) {
  __shared__ float sp[256], st[256];
  float a = 0.f, b = 0.f;
  for (int i = threadIdx.x; i < NUM_WG; i += 256) {
    a += partials[2 * i];
    b += partials[2 * i + 1];
  }
  sp[threadIdx.x] = a;
  st[threadIdx.x] = b;
  __syncthreads();
  for (int off = 128; off > 0; off >>= 1) {
    if (threadIdx.x < off) {
      sp[threadIdx.x] += sp[threadIdx.x + off];
      st[threadIdx.x] += st[threadIdx.x + off];
    }
    __syncthreads();
  }
  if (threadIdx.x == 0) {
    float l_pred = sp[0] / (float)(NYR * NPAP * LH);       // 750000
    float l_time = st[0] / (float)((T_SNAP - 1) * NPAP);   // 270000
    out[0] = l_pred + 1e-3f * l_time;
  }
}

extern "C" void kernel_launch(void* const* d_in, const int* in_sizes, int n_in,
                              void* d_out, int out_size, void* d_ws,
                              size_t ws_size, hipStream_t stream) {
  const float* X       = (const float*)d_in[0];
  const float* W_enc   = (const float*)d_in[1];
  const float* W_ih    = (const float*)d_in[2];
  const float* W_hh    = (const float*)d_in[3];
  const float* b_ih    = (const float*)d_in[4];
  const float* b_hh    = (const float*)d_in[5];
  const float* W_heads = (const float*)d_in[6];
  const float* b_heads = (const float*)d_in[7];
  const float* Y       = (const float*)d_in[8];
  const int*   years   = (const int*)d_in[9];

  char* ws = (char*)d_ws;
  float*     partials = (float*)(ws + WS_PARTIALS);
  _Float16*  wencF    = (_Float16*)(ws + WS_WENC);
  _Float16*  wihF     = (_Float16*)(ws + WS_WIH);
  _Float16*  whhF     = (_Float16*)(ws + WS_WHH);
  _Float16*  wheadsF  = (_Float16*)(ws + WS_WHEADS);

  prep_w<<<64, 32, 0, stream>>>(W_enc, wencF, 8, HDIM);      // 8 ktiles x 8 ntiles
  prep_w<<<96, 32, 0, stream>>>(W_ih, wihF, 24, 3 * HDIM);   // 4 x 24
  prep_w<<<96, 32, 0, stream>>>(W_hh, whhF, 24, 3 * HDIM);   // 4 x 24
  prep_w<<<4, 32, 0, stream>>>(W_heads, wheadsF, 1, 3);      // 4 x 1 (padded)

  impact_fused<<<NUM_WG, 256, 0, stream>>>(X, b_ih, b_hh, b_heads, Y, years,
                                           wencF, wihF, whhF, wheadsF,
                                           partials);
  finalize_loss<<<1, 256, 0, stream>>>(partials, (float*)d_out);
}
